// BiLSTM_20435454394481
// MI455X (gfx1250) — matrix operand order repaired and checked by reading, not compiled, not run on hardware
//
#include <hip/hip_runtime.h>
#include <math.h>

typedef __attribute__((ext_vector_type(16))) __bf16 v16bf;
typedef __attribute__((ext_vector_type(8)))  __bf16 v8bf;
typedef __attribute__((ext_vector_type(8)))  float  v8f;

#define TT   200   // time steps
#define BB   256   // batch
#define EE   300   // embed dim
#define HH   128   // hidden
#define KP   448   // padded K: [x(0..299) | zeros(300..319) | h(320..447)]
#define G4H  512   // 4*H gate width
#define XOFF 320   // where h lives inside an A row

__device__ __forceinline__ float sigf(float x) { return 1.0f / (1.0f + __expf(-x)); }

// ---------------------------------------------------------------------------
// Build transposed, padded bf16 weights: Wt[col][k], col in 0..511, k in 0..447.
// Source row map: k<300 -> W[k]; 300..319 -> 0; 320..447 -> W[k-20].
// N-major layout makes each WMMA B-fragment lane two contiguous 16B loads.
// ---------------------------------------------------------------------------
__global__ void prep_weights(const float* __restrict__ Wfw,
                             const float* __restrict__ Wbw,
                             __bf16* __restrict__ WtFw,
                             __bf16* __restrict__ WtBw) {
  int col = blockIdx.x;                       // 0..1023 (fw | bw)
  const float* W  = (col < G4H) ? Wfw  : Wbw;
  __bf16*      Wt = (col < G4H) ? WtFw : WtBw;
  int c = col & (G4H - 1);
  for (int k = threadIdx.x; k < KP; k += blockDim.x) {
    float v = 0.0f;
    if (k < EE)        v = W[(size_t)k * G4H + c];
    else if (k >= XOFF) v = W[(size_t)(k - 20) * G4H + c];
    Wt[(size_t)c * KP + k] = (__bf16)v;
  }
}

// ---------------------------------------------------------------------------
// Gather embeddings into time-major bf16 activation slices Xs[t*256+b][448].
// Cols 300..447 zeroed (320..447 is the h-slot filled by the previous step).
// ---------------------------------------------------------------------------
__global__ void gather_embed(const int* __restrict__ x,
                             const float* __restrict__ embeds,
                             __bf16* __restrict__ Xs) {
  int r = blockIdx.x;                         // r = t*256 + b
  int t = r >> 8;
  int b = r & 255;
  int row = x[b * TT + t];                    // x is [B][T]
  __bf16* dst = Xs + (size_t)r * KP;
  const float* src = embeds + (size_t)row * EE;
  for (int e = threadIdx.x; e < KP; e += blockDim.x)
    dst[e] = (__bf16)((e < EE) ? src[e] : 0.0f);
}

__global__ void zero_f32(float* p, int n) {
  int i = blockIdx.x * blockDim.x + threadIdx.x;
  if (i < n) p[i] = 0.0f;
}

// ---------------------------------------------------------------------------
// One fused LSTM step: z = A[256x448] @ Wt^T + b via v_wmma_f32_16x16x32_bf16,
// then gates + state update. 4 waves per block: wave g owns gate g's 16x16
// tile for this (M,N) block; tiles meet in LDS for the elementwise part.
// h is written (bf16) straight into the NEXT time slice's h-columns.
// ---------------------------------------------------------------------------
__global__ void __launch_bounds__(128)
lstm_step(const __bf16* __restrict__ A,      // [256][448] current slice
          const __bf16* __restrict__ Wt,     // [512][448] N-major weights
          const float*  __restrict__ bias,   // [512]
          const float*  __restrict__ c_in,   // [256][128]
          float*        __restrict__ c_out,  // [256][128]
          __bf16*       __restrict__ h_next, // next slice base, or nullptr
          float*        __restrict__ h_out)  // f32 h sink, or nullptr
{
  __shared__ float zT[4][16][17];            // +1 pad: no bank conflicts
  const int m0 = blockIdx.x * 16;            // batch-row tile
  const int n0 = blockIdx.y * 16;            // gate-local column tile (0..112)
  const int w  = threadIdx.x >> 5;           // wave id = gate id (i,j,f,o)
  const int l  = threadIdx.x & 31;
  const int ln = l & 15;
  const int lh = l >> 4;

  const int gcol = w * HH + n0 + ln;         // global z column of this lane
  const float bv = bias[gcol];
  v8f acc;
#pragma unroll
  for (int i = 0; i < 8; ++i) acc[i] = bv;   // bias folded into accumulator

  // A fragment (16x32 bf16): lane row = m0+ln; hi-lanes shift K by +8.
  const __bf16* arow = A  + (size_t)(m0 + ln) * KP + lh * 8;
  // B fragment (32x16 bf16): lane col = gcol; hi-lanes shift K by +8.
  const __bf16* brow = Wt + (size_t)gcol * KP + lh * 8;

#pragma unroll
  for (int k = 0; k < KP; k += 32) {
    v8bf a0 = *(const v8bf*)(arow + k);        // K + {0..7}  (+8 hi lanes)
    v8bf a1 = *(const v8bf*)(arow + k + 16);   // K + {16..23}(+8 hi lanes)
    v16bf av = __builtin_shufflevector(a0, a1, 0,1,2,3,4,5,6,7,8,9,10,11,12,13,14,15);
    v8bf b0 = *(const v8bf*)(brow + k);
    v8bf b1 = *(const v8bf*)(brow + k + 16);
    v16bf bw = __builtin_shufflevector(b0, b1, 0,1,2,3,4,5,6,7,8,9,10,11,12,13,14,15);
    acc = __builtin_amdgcn_wmma_f32_16x16x32_bf16(
        /*neg_a=*/false, av, /*neg_b=*/false, bw,
        /*c_mod=*/(short)0, acc, /*reuse_a=*/false, /*reuse_b=*/false);
  }

  // D layout: lane l holds (M = lh*8 + g, N = ln) in acc[g]
#pragma unroll
  for (int g = 0; g < 8; ++g)
    zT[w][lh * 8 + g][ln] = acc[g];
  __syncthreads();

  // gates: 256 elements / 128 threads
#pragma unroll
  for (int p = 0; p < 2; ++p) {
    int idx = threadIdx.x + p * 128;
    int rr = idx >> 4, nn = idx & 15;
    float zi = zT[0][rr][nn], zj = zT[1][rr][nn];
    float zf = zT[2][rr][nn], zo = zT[3][rr][nn];
    int bi = m0 + rr, ni = n0 + nn;
    float cp = c_in[bi * HH + ni];
    float cn = sigf(zf + 1.0f) * cp + sigf(zi) * tanhf(zj);
    float hv = sigf(zo) * tanhf(cn);
    c_out[bi * HH + ni] = cn;
    if (h_next) h_next[(size_t)bi * KP + XOFF + ni] = (__bf16)hv;
    if (h_out)  h_out[bi * HH + ni] = hv;
  }
}

// ---------------------------------------------------------------------------
// scores[b][c] = [h_fw | h_bw] @ w_out + b_out   (256x256x6 — negligible)
// ---------------------------------------------------------------------------
__global__ void out_proj(const float* __restrict__ hf,
                         const float* __restrict__ hb,
                         const float* __restrict__ w_out,
                         const float* __restrict__ b_out,
                         float* __restrict__ out) {
  int b = blockIdx.x;
  int c = threadIdx.x;
  if (c >= 6) return;
  float acc = b_out[c];
  for (int n = 0; n < HH; ++n) {
    acc += hf[b * HH + n] * w_out[n * 6 + c];
    acc += hb[b * HH + n] * w_out[(HH + n) * 6 + c];
  }
  out[b * 6 + c] = acc;
}

extern "C" void kernel_launch(void* const* d_in, const int* in_sizes, int n_in,
                              void* d_out, int out_size, void* d_ws, size_t ws_size,
                              hipStream_t stream) {
  const int*   x      = (const int*)d_in[0];
  const float* embeds = (const float*)d_in[1];
  const float* W_fw   = (const float*)d_in[2];
  const float* b_fw   = (const float*)d_in[3];
  const float* W_bw   = (const float*)d_in[4];
  const float* b_bw   = (const float*)d_in[5];
  const float* w_out  = (const float*)d_in[6];
  const float* b_out  = (const float*)d_in[7];
  float* out = (float*)d_out;

  char* ws = (char*)d_ws;
  size_t off = 0;
  auto carve = [&](size_t bytes) -> void* {
    void* p = ws + off;
    off += (bytes + 255) & ~(size_t)255;
    return p;
  };
  __bf16* Xs   = (__bf16*)carve(sizeof(__bf16) * (size_t)TT * BB * KP); // 45.9 MB
  __bf16* WtFw = (__bf16*)carve(sizeof(__bf16) * (size_t)G4H * KP);
  __bf16* WtBw = (__bf16*)carve(sizeof(__bf16) * (size_t)G4H * KP);
  float*  cF   = (float*) carve(sizeof(float) * BB * HH);
  float*  cB   = (float*) carve(sizeof(float) * BB * HH);
  float*  hF   = (float*) carve(sizeof(float) * BB * HH);
  float*  hB   = (float*) carve(sizeof(float) * BB * HH);
  (void)ws_size; (void)in_sizes; (void)n_in; (void)out_size;

  prep_weights<<<1024, 128, 0, stream>>>(W_fw, W_bw, WtFw, WtBw);
  gather_embed<<<TT * BB, 128, 0, stream>>>(x, embeds, Xs);
  zero_f32<<<(BB * HH + 255) / 256, 256, 0, stream>>>(cF, BB * HH);

  dim3 grid(BB / 16, HH / 16), blk(128);

  // Backward direction needs exactly ONE step on x[T-1] with zero state.
  // Run it first: slice T-1's h-columns are still zero from the gather,
  // and cF is still all-zero (it's read-only here).
  lstm_step<<<grid, blk, 0, stream>>>(Xs + (size_t)(TT - 1) * BB * KP,
                                      WtBw, b_bw, cF, cB, nullptr, hB);

  // Forward scan: 200 sequential fused steps (stream-ordered).
  for (int t = 0; t < TT; ++t) {
    __bf16* hn = (t + 1 < TT) ? Xs + (size_t)(t + 1) * BB * KP : nullptr;
    float*  ho = (t + 1 == TT) ? hF : nullptr;
    lstm_step<<<grid, blk, 0, stream>>>(Xs + (size_t)t * BB * KP,
                                        WtFw, b_fw, cF, cF, hn, ho);
  }

  out_proj<<<BB, 32, 0, stream>>>(hF, hB, w_out, b_out, out);
}
